// DirectFusionBlock_54674933678267
// MI455X (gfx1250) — matrix operand order repaired and checked by reading, not compiled, not run on hardware
//
#include <hip/hip_runtime.h>

typedef __attribute__((ext_vector_type(16))) _Float16 v16h;
typedef __attribute__((ext_vector_type(8)))  _Float16 v8h;
typedef __attribute__((ext_vector_type(8)))  float    v8f;
typedef __attribute__((ext_vector_type(4)))  int      i4v;

#define TILE 128
#define CC   128
#define HIDN 64
#define HFc  32
#define WFc  88

// f16 weight block layout (both in d_ws and in LDS):
//   [0, 32768)      We1  (256 x 128)
//   [32768, 49152)  We2  (128 x 128)
//   [49152, 57344)  W1fg (128 x 64)
#define WTOT 57344

#if defined(__has_builtin)
#if __has_builtin(__builtin_amdgcn_global_load_async_to_lds_b128)
#define HAS_ASYNC_LDS 1
#endif
#endif

// ---------------- prep: M_b = trans_b (3x4) @ inv(aug_b) (4x4) ----------------
__global__ void prep_mats(const float* __restrict__ aug, const float* __restrict__ trans,
                          float* __restrict__ mcomb, int B) {
    int b = threadIdx.x;
    if (b >= B) return;
    float A[4][4], I[4][4];
    for (int i = 0; i < 4; i++)
        for (int j = 0; j < 4; j++) {
            A[i][j] = aug[b * 16 + i * 4 + j];
            I[i][j] = (i == j) ? 1.0f : 0.0f;
        }
    for (int p = 0; p < 4; p++) {
        float inv = 1.0f / A[p][p];
        for (int j = 0; j < 4; j++) { A[p][j] *= inv; I[p][j] *= inv; }
        for (int r = 0; r < 4; r++) {
            if (r == p) continue;
            float f = A[r][p];
            for (int j = 0; j < 4; j++) { A[r][j] -= f * A[p][j]; I[r][j] -= f * I[p][j]; }
        }
    }
    for (int i = 0; i < 3; i++)
        for (int j = 0; j < 4; j++) {
            float s = 0.0f;
            for (int k = 0; k < 4; k++) s += trans[b * 12 + i * 4 + k] * I[k][j];
            mcomb[b * 12 + i * 4 + j] = s;
        }
}

// ---------------- prep: convert weights f32 -> f16 once, into d_ws ----------------
__global__ void prep_wts(const float* __restrict__ We1, const float* __restrict__ We2,
                         const float* __restrict__ W1fg, _Float16* __restrict__ wsW) {
    int i = blockIdx.x * 256 + threadIdx.x;
    if (i < 32768)      wsW[i] = (_Float16)We1[i];
    else if (i < 49152) wsW[i] = (_Float16)We2[i - 32768];
    else if (i < WTOT)  wsW[i] = (_Float16)W1fg[i - 49152];
}

// ---------------- WMMA fragment loaders (ISA 7.12.2 layouts) ----------------
// A 16x32 f16: lane l holds row m = l&15; halfs 0..7 -> K = k0..k0+7,
// halfs 8..15 -> K = k0+16..k0+23, where k0 = kBase + 8*(l>>4).
__device__ __forceinline__ v16h load_a_frag(const _Float16* base, int stride,
                                            int rowBase, int kBase, int lane) {
    const _Float16* p = base + (rowBase + (lane & 15)) * stride + kBase + ((lane >> 4) << 3);
    v8h lo = *(const v8h*)p;
    v8h hi = *(const v8h*)(p + 16);
    return __builtin_shufflevector(lo, hi, 0, 1, 2, 3, 4, 5, 6, 7, 8, 9, 10, 11, 12, 13, 14, 15);
}
// B 32x16 f16: lane l holds col N = nBase + (l&15); halfs 0..7 -> K = k0..k0+7,
// halfs 8..15 -> K = k0+16..k0+23, where k0 = kBase + 8*(l>>4).
__device__ __forceinline__ v16h load_b_frag(const _Float16* base, int stride,
                                            int kBase, int nBase, int lane) {
    const _Float16* p = base + (kBase + ((lane >> 4) << 3)) * stride + nBase + (lane & 15);
    v16h r;
    for (int s = 0; s < 8; s++) {
        r[s]     = p[s * stride];
        r[8 + s] = p[(16 + s) * stride];
    }
    return r;
}

#define WMMA(a, b, c) __builtin_amdgcn_wmma_f32_16x16x32_f16(false, (a), false, (b), (short)0, (c), false, false)

// ---------------- fused kernel ----------------
__global__ __launch_bounds__(256, 1) void fused_kernel(
    const float* __restrict__ x, const int* __restrict__ indices,
    const float* __restrict__ voxel_size, const float* __restrict__ pc_range,
    const float* __restrict__ img_feats,
    const _Float16* __restrict__ wsW,
    const float* __restrict__ b1fg, const float* __restrict__ W2fg,
    const float* __restrict__ b2fg,
    const float* __restrict__ be1, const float* __restrict__ be2,
    const float* __restrict__ gamma, const float* __restrict__ beta,
    const int* __restrict__ ihp, const int* __restrict__ iwp,
    const float* __restrict__ mcomb,
    float* __restrict__ out, int N) {

    __shared__ __align__(16) _Float16 sWt[WTOT];        // 112 KB: We1 | We2 | W1fg (f16)
    __shared__ __align__(16) _Float16 sA1[TILE * 256];  // 64 KB  [x | sampled] f16
    __shared__ __align__(16) _Float16 sH[TILE * CC];    // 32 KB  relu(GEMM1)
    __shared__ __align__(16) float    sOv[TILE * CC];   // 64 KB  fg-hidden, then delta
    __shared__ float sBW[TILE * 4];                     // bilinear weights
    __shared__ int   sBI[TILE * 4];                     // bilinear base offsets
    __shared__ float sValid[TILE];

    _Float16* sWe1 = sWt;
    _Float16* sWe2 = sWt + 32768;
    _Float16* sW1  = sWt + 49152;

    const int t    = threadIdx.x;
    const int lane = t & 31;
    const int wv   = t >> 5;
    const int rb   = wv * 16;            // this wave's 16-row strip
    const int row0 = blockIdx.x * TILE;

    // phase 0: stage f16 weights ws -> LDS.
    // Async path overlaps the 112 KB copy with geometry/gather below (ASYNCcnt).
#ifdef HAS_ASYNC_LDS
    {
        __attribute__((address_space(1))) i4v* g =
            (__attribute__((address_space(1))) i4v*)(unsigned long long)(uintptr_t)wsW;
        __attribute__((address_space(3))) i4v* l =
            (__attribute__((address_space(3))) i4v*)sWt;
        for (int i = t; i < (WTOT * 2) / 16; i += 256)
            __builtin_amdgcn_global_load_async_to_lds_b128(g + i, l + i, 0, 0);
    }
#else
    {
        const uint4* src = (const uint4*)wsW;
        uint4* dst = (uint4*)sWt;
        for (int i = t; i < (WTOT * 2) / 16; i += 256) dst[i] = src[i];
    }
#endif

    // phase 1: per-row geometry / projection
    if (t < TILE) {
        float w[4] = {0, 0, 0, 0};
        int bs[4] = {0, 0, 0, 0};
        float vf = 0.0f;
        int gm = row0 + t;
        if (gm < N) {
            int b  = indices[gm * 4 + 0];
            int zi = indices[gm * 4 + 1];
            int yi = indices[gm * 4 + 2];
            int xi = indices[gm * 4 + 3];
            float c0 = voxel_size[0] * 8.0f, c1 = voxel_size[1] * 8.0f, c2 = voxel_size[2] * 8.0f;
            float cx = (float)xi * c0 + pc_range[0] + 0.5f * c0;
            float cy = (float)yi * c1 + pc_range[1] + 0.5f * c1;
            float cz = (float)zi * c2 + pc_range[2] + 0.5f * c2;
            float rn = sqrtf(cx * cx + cy * cy) * (1.0f / 54.0f);
            rn = fminf(fmaxf(rn, 0.0f), 1.0f);
            const float* M = mcomb + b * 12;
            float p0 = M[0] * cx + M[1] * cy + M[2]  * cz + M[3];
            float p1 = M[4] * cx + M[5] * cy + M[6]  * cz + M[7];
            float p2 = M[8] * cx + M[9] * cy + M[10] * cz + M[11];
            float depth = p2;
            float sd = fmaxf(depth, 1e-5f);
            float u = p0 / sd, v = p1 / sd;
            float imw = (float)iwp[0], imh = (float)ihp[0];
            float un = 2.0f * (u * ((float)WFc / imw)) / (float)(WFc - 1) - 1.0f;
            float vn = 2.0f * (v * ((float)HFc / imh)) / (float)(HFc - 1) - 1.0f;
            bool valid = (depth > 1e-5f) && (fabsf(un) <= 1.0f) && (fabsf(vn) <= 1.0f);
            vf = valid ? 1.0f : 0.0f;
            float xp = (un + 1.0f) * 0.5f * (float)(WFc - 1);
            float yp = (vn + 1.0f) * 0.5f * (float)(HFc - 1);
            float x0 = floorf(xp), y0 = floorf(yp);
            float wx1 = xp - x0, wy1 = yp - y0;
            float cw[4]  = {(1 - wx1) * (1 - wy1), wx1 * (1 - wy1), (1 - wx1) * wy1, wx1 * wy1};
            float cxs[4] = {x0, x0 + 1, x0, x0 + 1};
            float cys[4] = {y0, y0, y0 + 1, y0 + 1};
            for (int j = 0; j < 4; j++) {
                bool ok = (cxs[j] >= 0.0f) && (cxs[j] <= (float)(WFc - 1)) &&
                          (cys[j] >= 0.0f) && (cys[j] <= (float)(HFc - 1));
                int xc = (int)fminf(fmaxf(cxs[j], 0.0f), (float)(WFc - 1));
                int yc = (int)fminf(fmaxf(cys[j], 0.0f), (float)(HFc - 1));
                w[j]  = ok ? cw[j] : 0.0f;
                bs[j] = b * CC * HFc * WFc + yc * WFc + xc;
            }
            out[(size_t)N * 130 + gm] = rn;
            out[(size_t)N * 131 + gm] = vf;
        }
        sValid[t] = vf;
        for (int j = 0; j < 4; j++) { sBW[t * 4 + j] = w[j]; sBI[t * 4 + j] = bs[j]; }
    }
    __syncthreads();

    // phase 2: build A1 = [x | bilinear(img)] in f16 (img_feats fits in L2)
    for (int e = t; e < TILE * CC; e += 256) {
        int row = e >> 7, c = e & 127;
        int gm = row0 + row;
        float xv = 0.0f, sv = 0.0f;
        if (gm < N) {
            xv = x[(size_t)gm * CC + c];
            int coff = c * HFc * WFc;
            sv = sBW[row * 4 + 0] * img_feats[sBI[row * 4 + 0] + coff] +
                 sBW[row * 4 + 1] * img_feats[sBI[row * 4 + 1] + coff] +
                 sBW[row * 4 + 2] * img_feats[sBI[row * 4 + 2] + coff] +
                 sBW[row * 4 + 3] * img_feats[sBI[row * 4 + 3] + coff];
        }
        sA1[row * 256 + c]       = (_Float16)xv;
        sA1[row * 256 + 128 + c] = (_Float16)sv;
    }

    // weights must be resident in LDS before any WMMA
#ifdef HAS_ASYNC_LDS
#if defined(__has_builtin) && __has_builtin(__builtin_amdgcn_s_wait_asynccnt)
    __builtin_amdgcn_s_wait_asynccnt(0);
#else
    asm volatile("s_wait_asynccnt 0" ::: "memory");
#endif
#endif
    __syncthreads();

    // phase 3: fg hidden = relu(x @ W1fg + b1fg), 16x64 per wave via WMMA
    // kt-outer / multi-accumulator: A loaded once per kt, WMMAs hit distinct accs.
    {
        v8f facc[4] = {};
        for (int kt = 0; kt < 4; kt++) {
            v16h a = load_a_frag(sA1, 256, rb, kt * 32, lane);
            for (int nt = 0; nt < 4; nt++) {
                v16h bm = load_b_frag(sW1, HIDN, kt * 32, nt * 16, lane);
                facc[nt] = WMMA(a, bm, facc[nt]);
            }
        }
        int mb = rb + ((lane >> 4) << 3);
        for (int nt = 0; nt < 4; nt++) {
            int n = nt * 16 + (lane & 15);
            float bias = b1fg[n];
            for (int r = 0; r < 8; r++)
                sOv[(mb + r) * HIDN + n] = fmaxf(facc[nt][r] + bias, 0.0f);
        }
    }
    __syncthreads();

    // phase 4: p_fg = sigmoid(hidden @ W2fg + b2fg)
    if (t < TILE) {
        int gm = row0 + t;
        if (gm < N) {
            float s = b2fg[0];
            for (int j = 0; j < HIDN; j++) s += sOv[t * HIDN + j] * W2fg[j];
            out[(size_t)N * 128 + gm] = 1.0f / (1.0f + expf(-s));
        }
    }
    __syncthreads();   // before sOv is reused as delta

    // phase 5: h = relu(A1 @ We1 + be1)  (K=256), 16x128 per wave
    {
        v8f acc[8] = {};
        for (int kt = 0; kt < 8; kt++) {
            v16h a = load_a_frag(sA1, 256, rb, kt * 32, lane);
            for (int nt = 0; nt < 8; nt++) {
                v16h bm = load_b_frag(sWe1, CC, kt * 32, nt * 16, lane);
                acc[nt] = WMMA(a, bm, acc[nt]);
            }
        }
        int mb = rb + ((lane >> 4) << 3);
        for (int nt = 0; nt < 8; nt++) {
            int n = nt * 16 + (lane & 15);
            float bias = be1[n];
            for (int r = 0; r < 8; r++)
                sH[(mb + r) * CC + n] = (_Float16)fmaxf(acc[nt][r] + bias, 0.0f);
        }
    }
    // sH rows rb..rb+15 are produced and consumed by this wave only;
    // per-wave LDS ordering (DScnt in-order) suffices, no barrier needed.

    // phase 6: delta = (h @ We2 + be2) * valid  (K=128)
    {
        v8f acc[8] = {};
        for (int kt = 0; kt < 4; kt++) {
            v16h a = load_a_frag(sH, CC, rb, kt * 32, lane);
            for (int nt = 0; nt < 8; nt++) {
                v16h bm = load_b_frag(sWe2, CC, kt * 32, nt * 16, lane);
                acc[nt] = WMMA(a, bm, acc[nt]);
            }
        }
        int mb = rb + ((lane >> 4) << 3);
        for (int nt = 0; nt < 8; nt++) {
            int n = nt * 16 + (lane & 15);
            float bias = be2[n];
            for (int r = 0; r < 8; r++) {
                int row = mb + r;
                sOv[row * CC + n] = (acc[nt][r] + bias) * sValid[row];
            }
        }
    }
    __syncthreads();

    // phase 7: epilogue — fusion gain + LayerNorm (one row per wave iteration)
    for (int i = 0; i < 16; i++) {
        int row = rb + i;
        int gm = row0 + row;
        if (gm >= N) continue;            // uniform across the wave
        float sd2 = 0, sx2 = 0, s1 = 0, s2 = 0;
        float xr[4];
        for (int q = 0; q < 4; q++) {
            int c = lane + q * 32;
            float dx = sOv[row * CC + c];
            float xx = x[(size_t)gm * CC + c];
            float r_ = xx + dx;
            xr[q] = r_;
            sd2 += dx * dx; sx2 += xx * xx; s1 += r_; s2 += r_ * r_;
        }
        for (int off = 16; off >= 1; off >>= 1) {
            sd2 += __shfl_down(sd2, off, 32);
            sx2 += __shfl_down(sx2, off, 32);
            s1  += __shfl_down(s1,  off, 32);
            s2  += __shfl_down(s2,  off, 32);
        }
        sd2 = __shfl(sd2, 0, 32); sx2 = __shfl(sx2, 0, 32);
        s1  = __shfl(s1,  0, 32); s2  = __shfl(s2,  0, 32);
        float mu   = s1 * (1.0f / 128.0f);
        float var  = s2 * (1.0f / 128.0f) - mu * mu;
        float rstd = rsqrtf(var + 1e-5f);
        float gain = sqrtf(sd2) / (sqrtf(sx2) + 1e-6f);
        float fg   = fminf(fmaxf(1.0f - expf(-gain), 0.0f), 1.0f);
        if (lane == 0) out[(size_t)N * 129 + gm] = fg;
        for (int q = 0; q < 4; q++) {
            int c = lane + q * 32;
            out[(size_t)gm * CC + c] = (xr[q] - mu) * rstd * gamma[c] + beta[c];
        }
    }
}

extern "C" void kernel_launch(void* const* d_in, const int* in_sizes, int n_in,
                              void* d_out, int out_size, void* d_ws, size_t ws_size,
                              hipStream_t stream) {
    const float* x          = (const float*)d_in[0];
    const int*   indices    = (const int*)d_in[1];
    const float* voxel_size = (const float*)d_in[2];
    const float* pc_range   = (const float*)d_in[3];
    const float* aug        = (const float*)d_in[4];
    const float* trans      = (const float*)d_in[5];
    const float* img        = (const float*)d_in[6];
    const float* W1fg       = (const float*)d_in[7];
    const float* b1fg       = (const float*)d_in[8];
    const float* W2fg       = (const float*)d_in[9];
    const float* b2fg       = (const float*)d_in[10];
    const float* We1        = (const float*)d_in[11];
    const float* be1        = (const float*)d_in[12];
    const float* We2        = (const float*)d_in[13];
    const float* be2        = (const float*)d_in[14];
    const float* gamma      = (const float*)d_in[15];
    const float* beta       = (const float*)d_in[16];
    const int*   ih         = (const int*)d_in[17];
    const int*   iw         = (const int*)d_in[18];

    int N = in_sizes[0] / CC;
    int B = in_sizes[4] / 16;

    // d_ws layout: [0,256) mcomb floats; [256, 256 + 114688) f16 weights
    float*     mcomb = (float*)d_ws;
    _Float16*  wsW   = (_Float16*)((char*)d_ws + 256);

    prep_mats<<<1, 32, 0, stream>>>(aug, trans, mcomb, B);
    prep_wts<<<(WTOT + 255) / 256, 256, 0, stream>>>(We1, We2, W1fg, wsW);

    int blocks = (N + TILE - 1) / TILE;
    fused_kernel<<<blocks, 256, 0, stream>>>(x, indices, voxel_size, pc_range, img,
                                             wsW, b1fg, W2fg, b2fg, be1, be2,
                                             gamma, beta, ih, iw, mcomb,
                                             (float*)d_out, N);
}
